// Attend_1726576857905
// MI455X (gfx1250) — matrix operand order repaired and checked by reading, not compiled, not run on hardware
//
#include <hip/hip_runtime.h>

typedef __attribute__((ext_vector_type(16))) _Float16 v16h;
typedef __attribute__((ext_vector_type(8)))  _Float16 v8h;
typedef __attribute__((ext_vector_type(4)))  _Float16 v4h;
typedef __attribute__((ext_vector_type(8)))  float    v8f;

#define LOG2E   1.44269504088896340736f
#define NEGBIG  (-1.0e38f)

constexpr int Nc = 2048;   // sequence length
constexpr int Hc = 16;     // heads
constexpr int Bc = 2;      // batch
constexpr int Dc = 64;     // head dim

static __device__ inline v8f wmma_f16(v16h a, v16h b, v8f c) {
  return __builtin_amdgcn_wmma_f32_16x16x32_f16(false, a, false, b, (short)0, c, false, false);
}

// A-fragment (16-bit, 16x32) from f32: halfs 0..15 <- row[base+0..15]
static __device__ inline v16h load_a_frag(const float* __restrict__ row, int base, float scale) {
  const float4* p0 = (const float4*)(row + base);
  const float4* p1 = (const float4*)(row + base + 8);
  float4 a = p0[0], b = p0[1], c = p1[0], d = p1[1];
  v16h f;
  f[0]=(_Float16)(a.x*scale); f[1]=(_Float16)(a.y*scale); f[2]=(_Float16)(a.z*scale); f[3]=(_Float16)(a.w*scale);
  f[4]=(_Float16)(b.x*scale); f[5]=(_Float16)(b.y*scale); f[6]=(_Float16)(b.z*scale); f[7]=(_Float16)(b.w*scale);
  f[8]=(_Float16)(c.x*scale); f[9]=(_Float16)(c.y*scale); f[10]=(_Float16)(c.z*scale); f[11]=(_Float16)(c.w*scale);
  f[12]=(_Float16)(d.x*scale); f[13]=(_Float16)(d.y*scale); f[14]=(_Float16)(d.z*scale); f[15]=(_Float16)(d.w*scale);
  return f;
}

// B-fragment (16-bit, 32x16) from f32: halfs 0..15 <- row[base+0..15]
static __device__ inline v16h load_b_frag_f32(const float* __restrict__ row, int base) {
  const float4* p = (const float4*)(row + base);
  float4 a = p[0], b = p[1], c = p[2], d = p[3];
  v16h f;
  f[0]=(_Float16)a.x;  f[1]=(_Float16)a.y;  f[2]=(_Float16)a.z;  f[3]=(_Float16)a.w;
  f[4]=(_Float16)b.x;  f[5]=(_Float16)b.y;  f[6]=(_Float16)b.z;  f[7]=(_Float16)b.w;
  f[8]=(_Float16)c.x;  f[9]=(_Float16)c.y;  f[10]=(_Float16)c.z; f[11]=(_Float16)c.w;
  f[12]=(_Float16)d.x; f[13]=(_Float16)d.y; f[14]=(_Float16)d.z; f[15]=(_Float16)d.w;
  return f;
}

// ---- pre-pass 1: K f32 -> f16, same row-major layout ----
__global__ __launch_bounds__(256)
void cvt_f16_kernel(const float* __restrict__ in, _Float16* __restrict__ out, int n4) {
  int i = blockIdx.x * 256 + threadIdx.x;
  if (i >= n4) return;
  float4 x = ((const float4*)in)[i];
  v4h o;
  o[0] = (_Float16)x.x; o[1] = (_Float16)x.y; o[2] = (_Float16)x.z; o[3] = (_Float16)x.w;
  ((v4h*)out)[i] = o;
}

// ---- pre-pass 2: V [bh][j][d] f32 -> Vt [bh][d][j] f16 (LDS-tiled transpose) ----
__global__ __launch_bounds__(256)
void transpose_v_kernel(const float* __restrict__ v, _Float16* __restrict__ vt) {
  __shared__ _Float16 tile[Dc][72];           // [d][j], padded stride
  const int tid = threadIdx.x;
  const int bh  = blockIdx.y;
  const int j0  = blockIdx.x * 64;
  const float* src = v + ((size_t)bh * Nc + j0) * Dc;
  #pragma unroll
  for (int it = 0; it < 4; ++it) {
    int idx = it * 256 + tid;
    int j = idx >> 4;                          // 0..63
    int d = (idx & 15) << 2;                   // 0,4,...,60
    float4 x = *(const float4*)(src + (size_t)j * Dc + d);
    tile[d + 0][j] = (_Float16)x.x;
    tile[d + 1][j] = (_Float16)x.y;
    tile[d + 2][j] = (_Float16)x.z;
    tile[d + 3][j] = (_Float16)x.w;
  }
  __syncthreads();
  #pragma unroll
  for (int it = 0; it < 2; ++it) {
    int idx = it * 256 + tid;
    int d  = idx >> 3;                         // 0..63
    int jc = (idx & 7) << 3;                   // 0,8,...,56
    v8h o;
    #pragma unroll
    for (int u = 0; u < 8; ++u) o[u] = tile[d][jc + u];
    *(v8h*)(vt + ((size_t)bh * Dc + d) * Nc + j0 + jc) = o;
  }
}

template <bool STAGED>
__global__ __launch_bounds__(256)
void fattn_kernel(const float* __restrict__ q, const float* __restrict__ k,
                  const float* __restrict__ v, const float* __restrict__ bias,
                  const unsigned char* __restrict__ mask,
                  const _Float16* __restrict__ kh, const _Float16* __restrict__ vt,
                  float* __restrict__ out) {
  __shared__ _Float16 plds[8][16 * 32];        // per-wave P staging (1 KB each)

  const int lane = threadIdx.x & 31;
  const int wave = threadIdx.x >> 5;
  const int t    = lane & 15;                  // column within 16-wide tile
  const int hi   = lane >> 4;                  // lane half (0/1)
  const int h    = blockIdx.y / Bc;            // b fastest-varying -> L2 bias reuse across batch
  const int b    = blockIdx.y % Bc;
  const int bh   = b * Hc + h;
  const int i0   = blockIdx.x * 128 + wave * 16;

  const float scale = 0.125f * LOG2E;          // 1/sqrt(64) * log2(e), folded into Q

  const float* Q  = q + (size_t)bh * Nc * Dc;
  const float* BI = bias + (size_t)h * Nc * Nc;
  const unsigned char* MK = mask + (size_t)b * Nc;
  float* O = out + (size_t)bh * Nc * Dc;

  // ---- Q fragments (A layout), pre-scaled ----
  const float* qrow = Q + (size_t)(i0 + t) * Dc;
  v16h qa0 = load_a_frag(qrow, 0 * 32 + hi * 16, scale);
  v16h qa1 = load_a_frag(qrow, 1 * 32 + hi * 16, scale);

  // ones B-fragment: accL += P @ ones gives exact f32 row sums of P
  v16h vones;
  #pragma unroll
  for (int i = 0; i < 16; ++i) vones[i] = (_Float16)1.0f;

  v8f acc0 = (v8f)0.0f, acc1 = (v8f)0.0f, acc2 = (v8f)0.0f, acc3 = (v8f)0.0f;
  v8f accL = (v8f)0.0f;

  _Float16* pw = &plds[wave][0];
  const int jmax = i0 + 15;                    // last valid (causal) column

  for (int j0 = 0; j0 <= jmax; j0 += 32) {
    // ---- K fragments (B layout: lane = key row, 16 contiguous d per frag) ----
    v16h kb00, kb01, kb10, kb11;
    if constexpr (STAGED) {
      const _Float16* kr0 = kh + (size_t)bh * Nc * Dc + (size_t)(j0 + t) * Dc;
      const _Float16* kr1 = kr0 + 16 * Dc;
      kb00 = *(const v16h*)(kr0 + hi * 16);
      kb01 = *(const v16h*)(kr0 + 32 + hi * 16);
      kb10 = *(const v16h*)(kr1 + hi * 16);
      kb11 = *(const v16h*)(kr1 + 32 + hi * 16);
    } else {
      const float* K = k + (size_t)bh * Nc * Dc;
      const float* krow0 = K + (size_t)(j0 + t) * Dc;
      const float* krow1 = K + (size_t)(j0 + 16 + t) * Dc;
      kb00 = load_b_frag_f32(krow0, hi * 16);
      kb01 = load_b_frag_f32(krow0, 32 + hi * 16);
      kb10 = load_b_frag_f32(krow1, hi * 16);
      kb11 = load_b_frag_f32(krow1, 32 + hi * 16);
    }

    // ---- S = (Q K^T) * scale * log2e ----
    v8f s0 = wmma_f16(qa0, kb00, (v8f)0.0f);
    s0     = wmma_f16(qa1, kb01, s0);
    v8f s1 = wmma_f16(qa0, kb10, (v8f)0.0f);
    s1     = wmma_f16(qa1, kb11, s1);

    // ---- bias + causal + key mask -> p = exp2(.) -> LDS as f16 ----
    const int  col0 = j0 + t;
    const int  col1 = col0 + 16;
    const bool edge = (j0 + 31 > i0);
    const bool mk0  = (MK[col0] != 0);
    const bool mk1  = (MK[col1] != 0);
    const float* brow = BI + (size_t)(i0 + 8 * hi) * Nc + j0 + t;
    #pragma unroll
    for (int r = 0; r < 8; ++r) {
      const int row = i0 + r + 8 * hi;
      float x0 = fmaf(brow[r * Nc],      LOG2E, s0[r]);
      float x1 = fmaf(brow[r * Nc + 16], LOG2E, s1[r]);
      if (edge) {
        if (col0 > row) x0 = NEGBIG;
        if (col1 > row) x1 = NEGBIG;
      }
      if (!mk0) x0 = NEGBIG;
      if (!mk1) x1 = NEGBIG;
      const float p0 = __builtin_amdgcn_exp2f(x0);
      const float p1 = __builtin_amdgcn_exp2f(x1);
      const int rowm = r + 8 * hi;
      pw[rowm * 32 + t]      = (_Float16)p0;   // C layout -> row-major LDS
      pw[rowm * 32 + 16 + t] = (_Float16)p1;
    }

    // ---- P: LDS -> A fragment (same-wave LDS ops are in-order) ----
    v8h plo = *(const v8h*)(pw + t * 32 + hi * 16);
    v8h phi = *(const v8h*)(pw + t * 32 + hi * 16 + 8);
    v16h pa;
    #pragma unroll
    for (int i = 0; i < 8; ++i) { pa[i] = plo[i]; pa[8 + i] = phi[i]; }

    // ---- row sums via WMMA ----
    accL = wmma_f16(pa, vones, accL);

    // ---- O += P V ----
    if constexpr (STAGED) {
      // Vt[bh][d][j]: per d-chunk one contiguous 32B load per lane
      const _Float16* vtp = vt + ((size_t)bh * Dc + t) * Nc + j0 + hi * 16;
      acc0 = wmma_f16(pa, *(const v16h*)(vtp + (size_t)0  * Nc), acc0);
      acc1 = wmma_f16(pa, *(const v16h*)(vtp + (size_t)16 * Nc), acc1);
      acc2 = wmma_f16(pa, *(const v16h*)(vtp + (size_t)32 * Nc), acc2);
      acc3 = wmma_f16(pa, *(const v16h*)(vtp + (size_t)48 * Nc), acc3);
    } else {
      const float* V  = v + (size_t)bh * Nc * Dc;
      const float* vp = V + (size_t)(j0 + hi * 16) * Dc + t;
      #pragma unroll
      for (int dt = 0; dt < 4; ++dt) {
        v16h vb;
        #pragma unroll
        for (int hh = 0; hh < 16; ++hh) vb[hh] = (_Float16)vp[(size_t)hh * Dc + dt * 16];
        if (dt == 0) acc0 = wmma_f16(pa, vb, acc0);
        if (dt == 1) acc1 = wmma_f16(pa, vb, acc1);
        if (dt == 2) acc2 = wmma_f16(pa, vb, acc2);
        if (dt == 3) acc3 = wmma_f16(pa, vb, acc3);
      }
    }
  }

  // ---- normalize + store ----
  #pragma unroll
  for (int r = 0; r < 8; ++r) {
    const float inv = 1.0f / accL[r];
    float* orow = O + (size_t)(i0 + r + 8 * hi) * Dc + t;
    orow[0]  = acc0[r] * inv;
    orow[16] = acc1[r] * inv;
    orow[32] = acc2[r] * inv;
    orow[48] = acc3[r] * inv;
  }
}

extern "C" void kernel_launch(void* const* d_in, const int* in_sizes, int n_in,
                              void* d_out, int out_size, void* d_ws, size_t ws_size,
                              hipStream_t stream) {
  const float*         q    = (const float*)d_in[0];
  const float*         k    = (const float*)d_in[1];
  const float*         v    = (const float*)d_in[2];
  const float*         bias = (const float*)d_in[3];
  const unsigned char* mask = (const unsigned char*)d_in[4];  // jnp bool -> 1 byte

  float* out = (float*)d_out;
  dim3 grid(Nc / 128, Bc * Hc);

  const size_t kvElems = (size_t)Bc * Hc * Nc * Dc;      // 4.19M
  const size_t need    = kvElems * 2 * sizeof(_Float16); // Kh + Vt = 16.8 MB

  if (ws_size >= need) {
    _Float16* kh = (_Float16*)d_ws;
    _Float16* vt = kh + kvElems;
    int n4 = (int)(kvElems / 4);
    hipLaunchKernelGGL(cvt_f16_kernel, dim3((n4 + 255) / 256), dim3(256), 0, stream, k, kh, n4);
    hipLaunchKernelGGL(transpose_v_kernel, dim3(Nc / 64, Bc * Hc), dim3(256), 0, stream, v, vt);
    hipLaunchKernelGGL((fattn_kernel<true>), grid, dim3(256), 0, stream,
                       q, k, v, bias, mask, kh, vt, out);
  } else {
    hipLaunchKernelGGL((fattn_kernel<false>), grid, dim3(256), 0, stream,
                       q, k, v, bias, mask, (const _Float16*)nullptr, (const _Float16*)nullptr, out);
  }
}
